// RoPEAttentionHead_39444979646581
// MI455X (gfx1250) — matrix operand order repaired and checked by reading, compiled, not run on hardware
//
#include <hip/hip_runtime.h>

#define D_MODEL 2048
#define SEQ     16
#define WAVES   8

typedef __attribute__((ext_vector_type(16))) __bf16         v16bf;
typedef __attribute__((ext_vector_type(8)))  float          v8f;
typedef __attribute__((ext_vector_type(16))) unsigned short us16;
typedef __attribute__((ext_vector_type(4)))  unsigned int   u32x4;

struct U32x8 { u32x4 lo, hi; };

__device__ __forceinline__ unsigned short f2bf(float f) {
  unsigned u = __builtin_bit_cast(unsigned, f);
  u += 0x7FFFu + ((u >> 16) & 1u);           // round-to-nearest-even
  return (unsigned short)(u >> 16);
}

__device__ __forceinline__ v16bf mkfrag(u32x4 a, u32x4 b) {
  U32x8 t; t.lo = a; t.hi = b;
  return __builtin_bit_cast(v16bf, t);
}

// A-fragment (16x32 bf16, contiguous k order, not the ISA table order): lane holds two 16B chunks of its row.
// lanes 0-15: K = k0+0..7 / k0+8..15 ; lanes 16-31: K = k0+16..23 / k0+24..31
__device__ __forceinline__ v16bf loadA(const unsigned short* base, int m, int k0, int hi8) {
  const unsigned short* p = base + m * D_MODEL + k0 + 2 * hi8;
  u32x4 a = *(const u32x4*)p;
  u32x4 b = *(const u32x4*)(p + 8);
  return mkfrag(a, b);
}

// Same, but from a 16x32 row-major scratch tile (stride 32)
__device__ __forceinline__ v16bf loadA32(const unsigned short* s, int lmod, int hi8) {
  const unsigned short* p = s + lmod * 32 + 2 * hi8;
  u32x4 a = *(const u32x4*)p;
  u32x4 b = *(const u32x4*)(p + 8);
  return mkfrag(a, b);
}

// B-fragment (32x16 bf16): lane n holds 16 contiguous K-values of its source row.
__device__ __forceinline__ v16bf loadB(const unsigned short* p) {
  u32x4 a = *(const u32x4*)p;
  u32x4 b = *(const u32x4*)(p + 8);
  return mkfrag(a, b);
}

__device__ __forceinline__ v8f wmma_bf16(v16bf A, v16bf B, v8f C) {
  return __builtin_amdgcn_wmma_f32_16x16x32_bf16(false, A, false, B, (short)0, C,
                                                 false, false);
}

// RoPE rotate one 16x16 accumulator tile in registers (columns live on lanes)
__device__ __forceinline__ v8f rope_reg(v8f acc, int col, int base_m) {
  const int   iidx    = col >> 1;
  // theta = 10000^(-2*(i-1)/2048), keeping the reference's (i-1) quirk
  const float theta   = __expf(-0.00899447309f * (float)(iidx - 1));
  const bool  evencol = (col & 1) == 0;
  v8f o;
#pragma unroll
  for (int r = 0; r < 8; ++r) {
    float own     = acc[r];
    float partner = __shfl_xor(own, 1, 32);
    float ang     = (float)(base_m + r) * theta;
    float c = __cosf(ang), s = __sinf(ang);
    o[r] = evencol ? (own * c + partner * s) : (own * c - partner * s);
  }
  return o;
}

// Store one 16x16 f32 accumulator tile as bf16 into a 16x32 scratch (stride 32)
__device__ __forceinline__ void store_tile(v8f a, unsigned short* s, int tcol,
                                           int lmod, int base_m) {
#pragma unroll
  for (int r = 0; r < 8; ++r)
    s[(base_m + r) * 32 + tcol + lmod] = f2bf(a[r]);
}

// attn @ v-tile: repack v accumulator into a B-fragment (K padded to 32), store out
__device__ __forceinline__ void av_store(v8f acc, v16bf attnA, bool hi, int lmod,
                                         int base_m, int n0, float* ob) {
  us16 tb;
#pragma unroll
  for (int j = 0; j < 16; ++j) tb[j] = 0;
#pragma unroll
  for (int r = 0; r < 8; ++r) {
    float hiv = __shfl(acc[r], lmod + 16, 32);
    tb[r]     = f2bf(hi ? hiv : acc[r]);
  }
  const v16bf Bv = __builtin_bit_cast(v16bf, tb);
  v8f oacc = {0.f, 0.f, 0.f, 0.f, 0.f, 0.f, 0.f, 0.f};
  oacc = wmma_bf16(attnA, Bv, oacc);
#pragma unroll
  for (int r = 0; r < 8; ++r)
    ob[(base_m + r) * D_MODEL + n0 + lmod] = oacc[r];
}

__global__ void cvt_bf16_kernel(const float* __restrict__ src,
                                unsigned short* __restrict__ dst, int n) {
  int i = blockIdx.x * blockDim.x + threadIdx.x;
  int stride = gridDim.x * blockDim.x;
  for (; i < n; i += stride) dst[i] = f2bf(src[i]);
}

// Two batch elements per workgroup: every weight B-fragment loaded from L2 is
// reused by both batches (2x less L2 weight traffic). Scores are accumulated
// on the fly (q_rot/k_rot never stored), keeping LDS at ~156 KB so two
// workgroups co-reside per WGP.
__launch_bounds__(256)
__global__ void rope_attn_kernel(const float* __restrict__ x,
                                 const unsigned short* __restrict__ wq,
                                 const unsigned short* __restrict__ wk,
                                 const unsigned short* __restrict__ wv,
                                 float* __restrict__ out) {
  extern __shared__ unsigned short smem[];
  unsigned short* xs   = smem;                   // 2 x 16 x 2048 bf16 = 128 KB
  unsigned short* wsc  = smem + 65536;           // 8 waves x 1536 (q0,q1,k) = 24 KB
  float* scoreacc      = (float*)(smem + 77824); // 2 x 16 x 16 f32 (2 KB)
  float* attnS         = scoreacc + 512;         // 2 x 16 x 16 f32 (2 KB)

  const int  b0     = blockIdx.x * 2;
  const int  tid    = threadIdx.x;
  const int  wave   = tid >> 5;
  const int  lane   = tid & 31;
  const int  lmod   = lane & 15;
  const bool hi     = lane >= 16;
  const int  hi8    = hi ? 8 : 0;
  const int  hi16   = hi ? 16 : 0;
  const int  base_m = hi ? 8 : 0;

  unsigned short* qs = wsc + wave * 1536;  // q scratch: [2][16][32]
  unsigned short* ks = qs + 1024;          // k scratch: [16][32]

  // ---- stage x for both batches (f32 -> bf16) + zero score accumulator ----
  for (int i = tid; i < 2 * SEQ * D_MODEL; i += 256) {
    int g   = i >> 15;
    int off = i & 32767;
    xs[i] = f2bf(x[(size_t)(b0 + g) * (SEQ * D_MODEL) + off]);
  }
  for (int i = tid; i < 512; i += 256) scoreacc[i] = 0.f;
  __syncthreads();

  const unsigned short* xs0 = xs;
  const unsigned short* xs1 = xs + 32768;
  const v8f vz = {0.f, 0.f, 0.f, 0.f, 0.f, 0.f, 0.f, 0.f};

  v8f sacc0 = vz, sacc1 = vz;  // partial scores per batch (this wave's e-cols)

  // ---- Q/K projections + RoPE + on-the-fly partial scores ----
  // wave owns 8 chunks of 32 e-columns (2 tiles each)
  for (int c = 0; c < 8; ++c) {
    const int e0 = c * 256 + wave * 32;

    // -- Q pass: 2 batches x 2 tiles, B fragments shared across batches --
    {
      const unsigned short* wr0 = wq + (size_t)(e0 +  0 + lmod) * D_MODEL + hi16;
      const unsigned short* wr1 = wq + (size_t)(e0 + 16 + lmod) * D_MODEL + hi16;
      v8f q00 = vz, q01 = vz, q10 = vz, q11 = vz;
      for (int k0 = 0; k0 < D_MODEL; k0 += 32) {
        v16bf A0 = loadA(xs0, lmod, k0, hi8);
        v16bf A1 = loadA(xs1, lmod, k0, hi8);
        v16bf B0 = loadB(wr0 + k0);
        v16bf B1 = loadB(wr1 + k0);
        q00 = wmma_bf16(A0, B0, q00);
        q01 = wmma_bf16(A0, B1, q01);
        q10 = wmma_bf16(A1, B0, q10);
        q11 = wmma_bf16(A1, B1, q11);
      }
      q00 = rope_reg(q00, e0 + lmod, base_m);
      q01 = rope_reg(q01, e0 + 16 + lmod, base_m);
      q10 = rope_reg(q10, e0 + lmod, base_m);
      q11 = rope_reg(q11, e0 + 16 + lmod, base_m);
      store_tile(q00, qs,        0, lmod, base_m);
      store_tile(q01, qs,       16, lmod, base_m);
      store_tile(q10, qs + 512,  0, lmod, base_m);
      store_tile(q11, qs + 512, 16, lmod, base_m);
    }

    // -- K pass --
    v8f k00 = vz, k01 = vz, k10 = vz, k11 = vz;
    {
      const unsigned short* wr0 = wk + (size_t)(e0 +  0 + lmod) * D_MODEL + hi16;
      const unsigned short* wr1 = wk + (size_t)(e0 + 16 + lmod) * D_MODEL + hi16;
      for (int k0 = 0; k0 < D_MODEL; k0 += 32) {
        v16bf A0 = loadA(xs0, lmod, k0, hi8);
        v16bf A1 = loadA(xs1, lmod, k0, hi8);
        v16bf B0 = loadB(wr0 + k0);
        v16bf B1 = loadB(wr1 + k0);
        k00 = wmma_bf16(A0, B0, k00);
        k01 = wmma_bf16(A0, B1, k01);
        k10 = wmma_bf16(A1, B0, k10);
        k11 = wmma_bf16(A1, B1, k11);
      }
      k00 = rope_reg(k00, e0 + lmod, base_m);
      k01 = rope_reg(k01, e0 + 16 + lmod, base_m);
      k10 = rope_reg(k10, e0 + lmod, base_m);
      k11 = rope_reg(k11, e0 + 16 + lmod, base_m);
    }

    // -- partial scores: sacc[g] += q_chunk[g] @ k_chunk[g]^T (K = 32 e-cols) --
    store_tile(k00, ks,  0, lmod, base_m);
    store_tile(k01, ks, 16, lmod, base_m);
    {
      v16bf A = loadA32(qs, lmod, hi8);
      v16bf B = loadB(ks + lmod * 32 + hi16);
      sacc0 = wmma_bf16(A, B, sacc0);
    }
    store_tile(k10, ks,  0, lmod, base_m);
    store_tile(k11, ks, 16, lmod, base_m);
    {
      v16bf A = loadA32(qs + 512, lmod, hi8);
      v16bf B = loadB(ks + lmod * 32 + hi16);
      sacc1 = wmma_bf16(A, B, sacc1);
    }
  }

  // ---- reduce partial scores across waves ----
#pragma unroll
  for (int r = 0; r < 8; ++r) {
    atomicAdd(&scoreacc[(base_m + r) * 16 + lmod], sacc0[r]);
    atomicAdd(&scoreacc[256 + (base_m + r) * 16 + lmod], sacc1[r]);
  }
  __syncthreads();

  // ---- softmax: wave g handles batch g (g = 0,1) ----
  if (wave < 2) {
    const int   g     = wave;
    const float scale = 0.02209708691f;  // 1/sqrt(2048)
#pragma unroll
    for (int r = 0; r < 8; ++r) {
      float v = scoreacc[g * 256 + (base_m + r) * 16 + lmod] * scale;
      float mx = v;
      for (int off = 1; off <= 8; off <<= 1) mx = fmaxf(mx, __shfl_xor(mx, off, 32));
      float e = __expf(v - mx);
      float sum = e;
      for (int off = 1; off <= 8; off <<= 1) sum += __shfl_xor(sum, off, 32);
      attnS[g * 256 + (base_m + r) * 16 + lmod] = e / sum;
    }
  }
  __syncthreads();

  // ---- per-wave attn A-fragments (16x32, K=16..31 zero padded), one per batch ----
  us16 ta0, ta1;
#pragma unroll
  for (int j = 0; j < 16; ++j) { ta0[j] = 0; ta1[j] = 0; }
#pragma unroll
  for (int j = 0; j < 8; ++j) {
    ta0[j] = f2bf(attnS[lmod * 16 + base_m + j]);
    ta1[j] = f2bf(attnS[256 + lmod * 16 + base_m + j]);
  }
  const v16bf attnA0 = __builtin_bit_cast(v16bf, ta0);
  const v16bf attnA1 = __builtin_bit_cast(v16bf, ta1);

  // ---- V projection + attn @ V + store (2 batches x 2 tiles) ----
  float* ob0 = out + (size_t)b0 * (SEQ * D_MODEL);
  float* ob1 = ob0 + SEQ * D_MODEL;
  for (int c = 0; c < 8; ++c) {
    const int n0 = c * 256 + wave * 32;
    const unsigned short* wr0 = wv + (size_t)(n0 +  0 + lmod) * D_MODEL + hi16;
    const unsigned short* wr1 = wv + (size_t)(n0 + 16 + lmod) * D_MODEL + hi16;
    v8f v00 = vz, v01 = vz, v10 = vz, v11 = vz;
    for (int k0 = 0; k0 < D_MODEL; k0 += 32) {
      v16bf A0 = loadA(xs0, lmod, k0, hi8);
      v16bf A1 = loadA(xs1, lmod, k0, hi8);
      v16bf B0 = loadB(wr0 + k0);
      v16bf B1 = loadB(wr1 + k0);
      v00 = wmma_bf16(A0, B0, v00);
      v01 = wmma_bf16(A0, B1, v01);
      v10 = wmma_bf16(A1, B0, v10);
      v11 = wmma_bf16(A1, B1, v11);
    }
    av_store(v00, attnA0, hi, lmod, base_m, n0,      ob0);
    av_store(v01, attnA0, hi, lmod, base_m, n0 + 16, ob0);
    av_store(v10, attnA1, hi, lmod, base_m, n0,      ob1);
    av_store(v11, attnA1, hi, lmod, base_m, n0 + 16, ob1);
  }
}

extern "C" void kernel_launch(void* const* d_in, const int* in_sizes, int n_in,
                              void* d_out, int out_size, void* d_ws, size_t ws_size,
                              hipStream_t stream) {
  const float* x  = (const float*)d_in[0];
  const float* wq = (const float*)d_in[1];
  const float* wk = (const float*)d_in[2];
  const float* wv = (const float*)d_in[3];
  float* out = (float*)d_out;

  const int welems = D_MODEL * D_MODEL; // 4,194,304
  unsigned short* wq_bf = (unsigned short*)d_ws;
  unsigned short* wk_bf = wq_bf + welems;
  unsigned short* wv_bf = wk_bf + welems;

  // one pass: f32 weights -> bf16 in workspace (24 MB, L2-resident afterwards)
  cvt_bf16_kernel<<<4096, 256, 0, stream>>>(wq, wq_bf, welems);
  cvt_bf16_kernel<<<4096, 256, 0, stream>>>(wk, wk_bf, welems);
  cvt_bf16_kernel<<<4096, 256, 0, stream>>>(wv, wv_bf, welems);

  // fused attention: one workgroup per PAIR of batch elements
  // LDS: 128K (x) + 24K (scratch) + 4K (scores/attn) = 159744 B -> 2 WGs/WGP
  const size_t smem_bytes = 159744;
  rope_attn_kernel<<<128, 256, smem_bytes, stream>>>(x, wq_bf, wk_bf, wv_bf, out);
}
